// Model_64768106824128
// MI455X (gfx1250) — matrix operand order repaired and checked by reading, not compiled, not run on hardware
//
#include <hip/hip_runtime.h>
#include <stdint.h>

// ---------------------------------------------------------------------------
// Graph ResNet forward for MI455X (gfx1250, wave32, WMMA).
// 30x [BN-folded GEMM (80000x256)x(256x128)] via v_wmma_f32_16x16x32_bf16
// with 32-row M-tiles (each B fragment feeds 2 WMMAs, halves L2 W traffic),
// SPMM scatter via global_atomic_add_f32, avg/stats/fold as small kernels.
// ---------------------------------------------------------------------------

#define NB   4
#define NN   20000
#define NE   160000
#define DD   128
#define ROWS (NB * NN)       // 80000
#define MT32 (NN / 32)       // 625 row tiles (32 rows) per graph
#define BN_EPS 1e-5f

typedef __attribute__((ext_vector_type(16))) __bf16 v16bf;
typedef __attribute__((ext_vector_type(8)))  __bf16 v8bf;
typedef __attribute__((ext_vector_type(8)))  float  v8f;

__device__ __forceinline__ float eluf(float v) {
    return v > 0.0f ? v : (__expf(v) - 1.0f);
}

__device__ __forceinline__ __bf16 f2bf(float f) {
    union { float f; uint32_t u; } x; x.f = f;
    uint32_t r = (x.u + 0x7FFFu + ((x.u >> 16) & 1u)) >> 16;  // RNE
    union { uint16_t u; __bf16 b; } y; y.u = (uint16_t)r;
    return y.b;
}

__device__ __forceinline__ void atomAddF(float* p, float v) {
    __hip_atomic_fetch_add(p, v, __ATOMIC_RELAXED, __HIP_MEMORY_SCOPE_AGENT);
}

// ---------------------------------------------------------------------------
__global__ void k_zero(float* p, int n) {
    int i = blockIdx.x * blockDim.x + threadIdx.x;
    int stride = gridDim.x * blockDim.x;
    for (; i < n; i += stride) p[i] = 0.0f;
}

// x = inputs(80000x6) @ W1(6x128) + b1
__global__ void k_conv1(const float* __restrict__ in6, const float* __restrict__ W1,
                        const float* __restrict__ b1, float* __restrict__ X) {
    int gid = blockIdx.x * blockDim.x + threadIdx.x;
    if (gid >= ROWS * DD) return;
    int r = gid / DD, c = gid % DD;
    float acc = b1[c];
    #pragma unroll
    for (int k = 0; k < 6; ++k) acc += in6[r * 6 + k] * W1[k * DD + c];
    X[gid] = acc;
}

// agg[b,row,:] += val * x[b,col,:]  (one edge per wave, 4 ch per lane)
__global__ void k_spmm(const int* __restrict__ rows, const int* __restrict__ cols,
                       const float* __restrict__ vals, const float* __restrict__ X,
                       float* __restrict__ AGG) {
    int tid  = blockIdx.x * blockDim.x + threadIdx.x;
    int eg   = tid >> 5;
    int lane = tid & 31;
    if (eg >= NB * NE) return;
    int b = eg / NE, e = eg - b * NE;
    int r = rows[b * NE + e];
    int c = cols[b * NE + e];
    float v = vals[b * NE + e];
    const float4* src = (const float4*)(X + ((size_t)b * NN + c) * DD);
    float*        dst = AGG + ((size_t)b * NN + r) * DD;
    float4 xv = src[lane];
    atomAddF(dst + lane * 4 + 0, v * xv.x);
    atomAddF(dst + lane * 4 + 1, v * xv.y);
    atomAddF(dst + lane * 4 + 2, v * xv.z);
    atomAddF(dst + lane * 4 + 3, v * xv.w);
}

// AVG[b,c] = sum_n x[b,n,c]*mask[b,n] / sum_n mask[b,n]   (block = (b,c))
__global__ void k_avg(const float* __restrict__ X, const float* __restrict__ mask,
                      float* __restrict__ AVG) {
    int b = blockIdx.x / DD, c = blockIdx.x % DD;
    int t = threadIdx.x;
    float s = 0.0f, m = 0.0f;
    for (int n = t; n < NN; n += 256) {
        float w = mask[b * NN + n];
        s += X[((size_t)b * NN + n) * DD + c] * w;
        m += w;
    }
    __shared__ float ss[256], sm[256];
    ss[t] = s; sm[t] = m;
    __syncthreads();
    for (int o = 128; o > 0; o >>= 1) {
        if (t < o) { ss[t] += ss[t + o]; sm[t] += sm[t + o]; }
        __syncthreads();
    }
    if (t == 0) AVG[b * DD + c] = ss[0] / sm[0];
}

// per-channel sum/sumsq of y = elu(concat(x, agg-or-avg)), 256 channels
__global__ void k_stats256(const float* __restrict__ X, const float* __restrict__ AG,
                           int isavg, float* __restrict__ stats) {
    int c = threadIdx.x;  // 0..255
    float s = 0.0f, s2 = 0.0f;
    for (int r = blockIdx.x; r < ROWS; r += gridDim.x) {
        float v;
        if (c < DD) v = X[(size_t)r * DD + c];
        else if (isavg) v = AG[(r / NN) * DD + (c - DD)];
        else v = AG[(size_t)r * DD + (c - DD)];
        v = eluf(v);
        s += v; s2 += v * v;
    }
    atomAddF(&stats[c], s);
    atomAddF(&stats[256 + c], s2);
}

// per-channel sum/sumsq of raw x, 128 channels (final BN)
__global__ void k_stats128(const float* __restrict__ X, float* __restrict__ stats) {
    int c = threadIdx.x;  // 0..127
    float s = 0.0f, s2 = 0.0f;
    for (int r = blockIdx.x; r < ROWS; r += gridDim.x) {
        float v = X[(size_t)r * DD + c];
        s += v; s2 += v * v;
    }
    atomAddF(&stats[c], s);
    atomAddF(&stats[DD + c], s2);
}

// a = gamma*rsqrt(var+eps); shift = beta - mu*a
__global__ void k_finalize(const float* __restrict__ stats, const float* __restrict__ gamma,
                           const float* __restrict__ beta, int C, float invM,
                           float* __restrict__ a, float* __restrict__ shift) {
    int c = threadIdx.x;
    if (c >= C) return;
    float mu  = stats[c] * invM;
    float var = stats[C + c] * invM - mu * mu;
    float rs  = rsqrtf(var + BN_EPS);
    float ac  = gamma[c] * rs;
    a[c] = ac;
    shift[c] = beta[c] - mu * ac;
}

// Wfrag[(kt*8+nt)*32+lane][h] = bf16(a[k] * W[k,n]), WMMA-B fragment order:
// half h <-> K = kt*32 + (h<8 ? h : h+8) + (lane<16 ? 0 : 8), N = nt*16 + lane%16
__global__ void k_foldW(const float* __restrict__ a, const float* __restrict__ W,
                        __bf16* __restrict__ Wfrag) {
    int kt = blockIdx.x;            // 0..7
    int idx = threadIdx.x;          // 0..255
    int nt = idx >> 5, lane = idx & 31;
    int n = nt * 16 + (lane & 15);
    int kofs = (lane < 16) ? 0 : 8;
    __bf16* out = Wfrag + (((kt * 8 + nt) * 32 + lane) << 4);
    #pragma unroll
    for (int h = 0; h < 16; ++h) {
        int k = kt * 32 + (h < 8 ? h : h + 8) + kofs;
        out[h] = f2bf(a[k] * W[k * DD + n]);
    }
}

// bfused[n] = b[n] + sum_k shift[k]*W[k,n]
__global__ void k_foldB(const float* __restrict__ shift, const float* __restrict__ W,
                        const float* __restrict__ bvec, int C, float* __restrict__ bf) {
    int n = threadIdx.x;  // 0..127
    float s = bvec[n];
    for (int k = 0; k < C; ++k) s += shift[k] * W[k * DD + n];
    bf[n] = s;
}

// GEMM: XO[32 rows x 128] = elu(concat(x,agg)) tile @ Wfrag + bfused (+resid)
// 8 waves/block; each wave owns one 16-col N slab over a 32-row M tile:
// per k-step, one B fragment feeds two WMMAs (row-halves of the M tile).
__global__ void __launch_bounds__(256)
k_gemm(const float* __restrict__ X, const float* __restrict__ AGG,
       const float* __restrict__ AVG, int isavg,
       const __bf16* __restrict__ Wfrag, const float* __restrict__ bfused,
       const float* __restrict__ resid, float* __restrict__ XO) {
    __shared__ __bf16 yl[32 * 256];   // 16 KB activation tile (bf16)

    int b = blockIdx.x / MT32;
    int m0 = (blockIdx.x % MT32) * 32;
    int t = threadIdx.x;

    // Phase 1: build y tile = elu(concat(x, agg)) in LDS (2 tasks/thread)
    #pragma unroll
    for (int it = 0; it < 2; ++it) {
        int task = t + it * 256;                 // 0..511
        int row = task >> 4, seg = task & 15;    // seg: 16-channel chunk
        size_t gr = (size_t)b * NN + m0 + row;
        __bf16* dst = yl + row * 256 + seg * 16;
        if (seg < 8) {
            const float4* p = (const float4*)(X + gr * DD + seg * 16);
            #pragma unroll
            for (int q = 0; q < 4; ++q) {
                float4 v = p[q];
                dst[q * 4 + 0] = f2bf(eluf(v.x));
                dst[q * 4 + 1] = f2bf(eluf(v.y));
                dst[q * 4 + 2] = f2bf(eluf(v.z));
                dst[q * 4 + 3] = f2bf(eluf(v.w));
            }
        } else if (isavg) {
            const float* p = AVG + b * DD + (seg - 8) * 16;
            #pragma unroll
            for (int q = 0; q < 16; ++q) dst[q] = f2bf(eluf(p[q]));
        } else {
            const float4* p = (const float4*)(AGG + gr * DD + (seg - 8) * 16);
            #pragma unroll
            for (int q = 0; q < 4; ++q) {
                float4 v = p[q];
                dst[q * 4 + 0] = f2bf(eluf(v.x));
                dst[q * 4 + 1] = f2bf(eluf(v.y));
                dst[q * 4 + 2] = f2bf(eluf(v.z));
                dst[q * 4 + 3] = f2bf(eluf(v.w));
            }
        }
    }
    __syncthreads();

    // Phase 2: WMMA over K; two row-halves share each B fragment
    int wave = t >> 5, lane = t & 31;
    int mrow = lane & 15;
    const __bf16* arow0 = yl + mrow * 256 + ((lane < 16) ? 0 : 8);
    const __bf16* arow1 = arow0 + 16 * 256;
    v8f acc0 = {};
    v8f acc1 = {};
    #pragma unroll
    for (int kt = 0; kt < 8; ++kt) {
        // B fragment: pre-swizzled contiguous 32 B per lane (L2-resident)
        v16bf bv = *(const v16bf*)(Wfrag + (((kt * 8 + wave) * 32 + lane) << 4));
        // A fragments: halves 0..7 = K {base..base+7}, halves 8..15 = +16
        const __bf16* ap0 = arow0 + kt * 32;
        v8bf lo0 = *(const v8bf*)ap0;
        v8bf hi0 = *(const v8bf*)(ap0 + 16);
        const __bf16* ap1 = arow1 + kt * 32;
        v8bf lo1 = *(const v8bf*)ap1;
        v8bf hi1 = *(const v8bf*)(ap1 + 16);
        v16bf av0, av1;
        #pragma unroll
        for (int q = 0; q < 8; ++q) {
            av0[q] = lo0[q]; av0[q + 8] = hi0[q];
            av1[q] = lo1[q]; av1[q + 8] = hi1[q];
        }
        acc0 = __builtin_amdgcn_wmma_f32_16x16x32_bf16(
                   false, av0, false, bv, (short)0, acc0, false, false);
        acc1 = __builtin_amdgcn_wmma_f32_16x16x32_bf16(
                   false, av1, false, bv, (short)0, acc1, false, false);
    }

    // Phase 3: epilogue — bias (+ residual), store f32
    int n = wave * 16 + (lane & 15);
    int mbase = (lane < 16) ? 0 : 8;
    float bn = bfused[n];
    #pragma unroll
    for (int r = 0; r < 8; ++r) {
        size_t oi0 = ((size_t)b * NN + m0 + mbase + r) * DD + n;
        float v0 = acc0[r] + bn;
        if (resid) v0 += resid[oi0];
        XO[oi0] = v0;
        size_t oi1 = oi0 + (size_t)16 * DD;
        float v1 = acc1[r] + bn;
        if (resid) v1 += resid[oi1];
        XO[oi1] = v1;
    }
}

// out[r] = elu( sum_k (a[k]*x[r,k]+shift[k]) * W2[k] + b2 )
__global__ void k_final(const float* __restrict__ X, const float* __restrict__ a,
                        const float* __restrict__ shift, const float* __restrict__ W2,
                        const float* __restrict__ b2, float* __restrict__ out) {
    int r = blockIdx.x * blockDim.x + threadIdx.x;
    if (r >= ROWS) return;
    float s = b2[0];
    #pragma unroll 4
    for (int k = 0; k < DD; ++k)
        s += (a[k] * X[(size_t)r * DD + k] + shift[k]) * W2[k];
    out[r] = eluf(s);
}

// ---------------------------------------------------------------------------
extern "C" void kernel_launch(void* const* d_in, const int* in_sizes, int n_in,
                              void* d_out, int out_size, void* d_ws, size_t ws_size,
                              hipStream_t stream) {
    const float* inputs = (const float*)d_in[0];
    const float* mask   = (const float*)d_in[1];
    const int*   Lrows  = (const int*)  d_in[2];
    const int*   Lcols  = (const int*)  d_in[3];
    const float* Lvals  = (const float*)d_in[4];
    const float* W1     = (const float*)d_in[5];
    const float* b1     = (const float*)d_in[6];
    const float* rng    = (const float*)d_in[7];   // [15,2,256]
    const float* rnb    = (const float*)d_in[8];   // [15,2,256]
    const float* rnW    = (const float*)d_in[9];   // [15,2,256,128]
    const float* rnbias = (const float*)d_in[10];  // [15,2,128]
    const float* g2     = (const float*)d_in[11];
    const float* be2    = (const float*)d_in[12];
    const float* W2     = (const float*)d_in[13];
    const float* b2     = (const float*)d_in[14];
    float* out = (float*)d_out;

    // workspace layout (all chunks 256-B aligned)
    const size_t S = (size_t)ROWS * DD * sizeof(float);   // 40.96 MB
    char* w = (char*)d_ws;
    float*  X0    = (float*)w;            w += S;
    float*  X1    = (float*)w;            w += S;
    float*  AGG   = (float*)w;            w += S;
    float*  AVG   = (float*)w;            w += NB * DD * sizeof(float);
    float*  STATS = (float*)w;            w += 512 * sizeof(float);
    float*  AV    = (float*)w;            w += 256 * sizeof(float);
    float*  SH    = (float*)w;            w += 256 * sizeof(float);
    float*  BF    = (float*)w;            w += 128 * sizeof(float);
    __bf16* WF    = (__bf16*)w;           w += 256 * 128 * sizeof(__bf16);

    const float invM = 1.0f / (float)ROWS;

    k_conv1<<<(ROWS * DD + 255) / 256, 256, 0, stream>>>(inputs, W1, b1, X0);

    for (int i = 0; i < 15; ++i) {
        for (int j = 0; j < 2; ++j) {
            const float* xin   = j ? X1 : X0;
            float*       xout  = j ? X0 : X1;
            const float* resid = j ? X0 : nullptr;
            int isavg = (i & 1);
            int s = i * 2 + j;

            if (!isavg) {
                k_zero<<<4096, 256, 0, stream>>>(AGG, ROWS * DD);
                k_spmm<<<(NB * NE * 32) / 256, 256, 0, stream>>>(
                    Lrows, Lcols, Lvals, xin, AGG);
            } else {
                k_avg<<<NB * DD, 256, 0, stream>>>(xin, mask, AVG);
            }

            k_zero<<<2, 256, 0, stream>>>(STATS, 512);
            k_stats256<<<256, 256, 0, stream>>>(xin, isavg ? AVG : AGG, isavg, STATS);
            k_finalize<<<1, 256, 0, stream>>>(STATS, rng + s * 256, rnb + s * 256,
                                              256, invM, AV, SH);
            k_foldW<<<8, 256, 0, stream>>>(AV, rnW + (size_t)s * 256 * DD, WF);
            k_foldB<<<1, 128, 0, stream>>>(SH, rnW + (size_t)s * 256 * DD,
                                           rnbias + s * DD, 256, BF);
            k_gemm<<<NB * MT32, 256, 0, stream>>>(xin, AGG, AVG, isavg, WF, BF,
                                                  resid, xout);
        }
    }

    // final BN + 128->1 conv + elu
    k_zero<<<1, 512, 0, stream>>>(STATS, 512);
    k_stats128<<<256, 128, 0, stream>>>(X0, STATS);
    k_finalize<<<1, 128, 0, stream>>>(STATS, g2, be2, 128, invM, AV, SH);
    k_final<<<(ROWS + 255) / 256, 256, 0, stream>>>(X0, AV, SH, W2, b2, out);

    (void)in_sizes; (void)n_in; (void)out_size; (void)ws_size;
}
